// LSHNNetwork_84808424227419
// MI455X (gfx1250) — matrix operand order repaired and checked by reading, not compiled, not run on hardware
//
#include <hip/hip_runtime.h>
#include <hip/hip_bf16.h>
#include <math.h>

// ---------------------------------------------------------------------------
// LSHN retrieval: cosine-sim GEMM (f16 WMMA) + chunked top-32 + softmax gather
// Sizes fixed by reference: B=256, C=262144, P=512, K=32.
// ---------------------------------------------------------------------------

#define BQ   256
#define CPAT 262144
#define PD   512
#define TOPK 32
#define CCHUNK 16384
#define NEG_BIG (-3.402823466e38f)

typedef _Float16 v16h __attribute__((ext_vector_type(16)));
typedef _Float16 v8h  __attribute__((ext_vector_type(8)));
typedef float    v8f  __attribute__((ext_vector_type(8)));

// ---------------------------------------------------------------------------
// Kernel 1: normalize queries -> f16, and init running top-k buffers.
// grid = (256), block = (256)
// ---------------------------------------------------------------------------
__global__ void lshn_qnorm(const float* __restrict__ q, _Float16* __restrict__ qn,
                           float* __restrict__ rscore, int* __restrict__ ridx) {
    const int b = blockIdx.x, t = threadIdx.x;
    __shared__ float red[256];
    const float x0 = q[b * PD + t];
    const float x1 = q[b * PD + 256 + t];
    red[t] = x0 * x0 + x1 * x1;
    __syncthreads();
    for (int off = 128; off > 0; off >>= 1) {
        if (t < off) red[t] += red[t + off];
        __syncthreads();
    }
    const float inv = 1.0f / fmaxf(sqrtf(red[0]), 1e-8f);
    qn[b * PD + t]       = (_Float16)(x0 * inv);
    qn[b * PD + 256 + t] = (_Float16)(x1 * inv);
    if (t < TOPK) {
        rscore[b * TOPK + t] = NEG_BIG;
        ridx[b * TOPK + t]   = 0;
    }
}

// ---------------------------------------------------------------------------
// Kernel 2: per-pattern inverse norms (one wave32 per pattern row).
// grid = (C/8), block = (256)
// ---------------------------------------------------------------------------
__global__ void lshn_pnorm(const float* __restrict__ pat, float* __restrict__ invp) {
    const int wave = threadIdx.x >> 5, lane = threadIdx.x & 31;
    const int row = blockIdx.x * 8 + wave;
    const float4* pr = (const float4*)(pat + (size_t)row * PD + lane * 16);
    float s = 0.f;
#pragma unroll
    for (int i = 0; i < 4; ++i) {
        float4 v = pr[i];
        s += v.x * v.x + v.y * v.y + v.z * v.z + v.w * v.w;
    }
#pragma unroll
    for (int m = 16; m > 0; m >>= 1) s += __shfl_xor(s, m, 32);
    if (lane == 0) invp[row] = 1.0f / fmaxf(sqrtf(s), 1e-8f);
}

// ---------------------------------------------------------------------------
// Kernel 3: sims chunk GEMM, f16 WMMA with f32 accumulate.
// grid = (CCHUNK/32, 2), block = (256) = 8 waves.
// Wave w: M rows [blockIdx.y*128 + w*16, +16), N cols [blockIdx.x*32, +32)
// as TWO 16x16 tiles sharing one A fragment (two independent accumulator
// chains -> conversions of one tile fill the other's WMMA hazard window).
// sims[b][n] = (qn[b,:] . pat[active n,:]) * inv_pnorm[active n]
// ---------------------------------------------------------------------------
__global__ void lshn_gemm(const float* __restrict__ pat, const _Float16* __restrict__ qn,
                          const float* __restrict__ invp, const int* __restrict__ wptr,
                          float* __restrict__ sims, int chunk_base) {
    const int wave = threadIdx.x >> 5;
    const int lane = threadIdx.x & 31;
    const int g  = lane >> 4;      // k-group for A/B fragments, row-group for D
    const int nl = lane & 15;      // A: local m, B/D: local n
    const int ntile = blockIdx.x * 32;
    const int mbase = blockIdx.y * 128 + wave * 16;
    const int W = *wptr;

    // B-matrix source rows for this lane (active index -> circular buffer row)
    const int r0 = (chunk_base + ntile + nl + W) & (CPAT - 1);
    const int r1 = (chunk_base + ntile + 16 + nl + W) & (CPAT - 1);
    const float* prow0 = pat + (size_t)r0 * PD;
    const float* prow1 = pat + (size_t)r1 * PD;
    const _Float16* qrow = qn + (size_t)(mbase + nl) * PD;

    v8f acc0 = {};
    v8f acc1 = {};
#pragma unroll
    for (int kb = 0; kb < PD; kb += 32) {
        // A fragment (16x32 f16): half h <-> k = kb + g*16 + h
        union { v8h h8[2]; v16h h16; } au;
        au.h8[0] = *(const v8h*)(qrow + kb + g * 16);
        au.h8[1] = *(const v8h*)(qrow + kb + g * 16 + 8);

        // B fragments (32x16 f16): half h <-> k = kb + g*16 + h, col = nl
        const float4* bp0 = (const float4*)(prow0 + kb + g * 16);
        const float4* bp1 = (const float4*)(prow1 + kb + g * 16);
        float f0[16], f1[16];
        *(float4*)(f0 + 0)  = bp0[0];
        *(float4*)(f0 + 4)  = bp0[1];
        *(float4*)(f0 + 8)  = bp0[2];
        *(float4*)(f0 + 12) = bp0[3];
        *(float4*)(f1 + 0)  = bp1[0];
        *(float4*)(f1 + 4)  = bp1[1];
        *(float4*)(f1 + 8)  = bp1[2];
        *(float4*)(f1 + 12) = bp1[3];
        v16h bv0, bv1;
#pragma unroll
        for (int i = 0; i < 16; ++i) bv0[i] = (_Float16)f0[i];
#pragma unroll
        for (int i = 0; i < 16; ++i) bv1[i] = (_Float16)f1[i];

        if (kb + 32 < PD) {
            __builtin_prefetch(prow0 + kb + 32 + g * 16, 0, 3);  // global_prefetch_b8
            __builtin_prefetch(prow1 + kb + 32 + g * 16, 0, 3);
        }

        acc0 = __builtin_amdgcn_wmma_f32_16x16x32_f16(
            false, au.h16, false, bv0, (short)0, acc0, false, false);
        acc1 = __builtin_amdgcn_wmma_f32_16x16x32_f16(
            false, au.h16, false, bv1, (short)0, acc1, false, false);
    }

    // Epilogue: fold per-pattern inverse norm; D layout: lane n = nl, m = g*8+r
    const float s0 = invp[r0];
    const float s1 = invp[r1];
#pragma unroll
    for (int r = 0; r < 8; ++r) {
        const int brow = mbase + g * 8 + r;
        sims[(size_t)brow * CCHUNK + ntile + nl]      = acc0[r] * s0;
        sims[(size_t)brow * CCHUNK + ntile + 16 + nl] = acc1[r] * s1;
    }
}

// ---------------------------------------------------------------------------
// Kernel 4: merge running top-32 with this chunk's 16384 scores per query.
// grid = (256) one block per query, block = (256). Iterative max-extraction
// (32 rounds) yields scores sorted descending, matching jax.lax.top_k.
// ---------------------------------------------------------------------------
__global__ void lshn_topk(const float* __restrict__ sims, float* __restrict__ rscore,
                          int* __restrict__ ridx, int chunk_base) {
    const int b = blockIdx.x, t = threadIdx.x;
    float v[64];
    const float* row = sims + (size_t)b * CCHUNK + t * 64;
#pragma unroll
    for (int i = 0; i < 64; ++i) v[i] = row[i];

    float extra   = (t < TOPK) ? rscore[b * TOPK + t] : NEG_BIG;
    int   extraid = (t < TOPK) ? ridx[b * TOPK + t] : 0;

    __shared__ float sval[256];
    __shared__ int   sgid[256];
    __shared__ int   sown[256];
    __shared__ float outv[TOPK];
    __shared__ int   outi[TOPK];

    for (int iter = 0; iter < TOPK; ++iter) {
        float best = NEG_BIG;
        int slot = 0;
        for (int i = 0; i < 64; ++i)
            if (v[i] > best) { best = v[i]; slot = i; }
        if (extra > best) { best = extra; slot = 64; }
        const int gidx = (slot == 64) ? extraid : (chunk_base + t * 64 + slot);

        sval[t] = best;
        sgid[t] = gidx;
        sown[t] = (t << 8) | slot;
        __syncthreads();
        for (int off = 128; off > 0; off >>= 1) {
            if (t < off && sval[t + off] > sval[t]) {
                sval[t] = sval[t + off];
                sgid[t] = sgid[t + off];
                sown[t] = sown[t + off];
            }
            __syncthreads();
        }
        const int wown = sown[0];
        if (t == 0) { outv[iter] = sval[0]; outi[iter] = sgid[0]; }
        if (t == (wown >> 8)) {
            const int sl = wown & 255;
            if (sl == 64) extra = NEG_BIG;
            else          v[sl] = NEG_BIG;
        }
        __syncthreads();
    }
    if (t < TOPK) {
        rscore[b * TOPK + t] = outv[t];
        ridx[b * TOPK + t]   = outi[t];
    }
}

// ---------------------------------------------------------------------------
// Kernel 5: softmax weights over top-32 + weighted gather; emit outputs.
// d_out = retrieved[256][512] then top_scores[256][32].
// grid = (256), block = (256)
// ---------------------------------------------------------------------------
__global__ void lshn_aggregate(const float* __restrict__ pat, const float* __restrict__ rscore,
                               const int* __restrict__ ridx, const int* __restrict__ wptr,
                               float* __restrict__ out) {
    const int b = blockIdx.x, t = threadIdx.x;
    __shared__ float w[TOPK];
    __shared__ int   rows[TOPK];
    __shared__ float ssum;
    const int W = *wptr;
    if (t < TOPK) {
        const float s0 = rscore[b * TOPK + 0];  // sorted descending -> max
        w[t] = expf(rscore[b * TOPK + t] - s0);
        rows[t] = (ridx[b * TOPK + t] + W) & (CPAT - 1);
    }
    __syncthreads();
    if (t == 0) {
        float s = 0.f;
        for (int k = 0; k < TOPK; ++k) s += w[k];
        ssum = s;
    }
    __syncthreads();

    float a0 = 0.f, a1 = 0.f;
#pragma unroll 4
    for (int k = 0; k < TOPK; ++k) {
        const float* pr = pat + (size_t)rows[k] * PD;
        a0 += w[k] * pr[t];
        a1 += w[k] * pr[t + 256];
    }
    const float inv = 1.0f / ssum;
    out[b * PD + t]       = a0 * inv;
    out[b * PD + 256 + t] = a1 * inv;
    if (t < TOPK) out[BQ * PD + b * TOPK + t] = rscore[b * TOPK + t];
}

// ---------------------------------------------------------------------------
extern "C" void kernel_launch(void* const* d_in, const int* in_sizes, int n_in,
                              void* d_out, int out_size, void* d_ws, size_t ws_size,
                              hipStream_t stream) {
    const float* q    = (const float*)d_in[0];   // [256, 512]
    const float* pat  = (const float*)d_in[1];   // [262144, 512]
    const int*   wptr = (const int*)d_in[2];     // write_index (scalar)
    // d_in[3] = top_k = 32 (compile-time constant here)
    float* out = (float*)d_out;

    char* ws = (char*)d_ws;
    size_t off = 0;
    _Float16* qn   = (_Float16*)(ws + off); off += (size_t)BQ * PD * sizeof(_Float16);  // 256 KB
    float* invp    = (float*)(ws + off);    off += (size_t)CPAT * sizeof(float);        // 1 MB
    float* rscore  = (float*)(ws + off);    off += (size_t)BQ * TOPK * sizeof(float);
    int*   ridx    = (int*)(ws + off);      off += (size_t)BQ * TOPK * sizeof(int);
    float* sims    = (float*)(ws + off);    // 256 * 16384 * 4 = 16 MB chunk tile

    lshn_qnorm<<<BQ, 256, 0, stream>>>(q, qn, rscore, ridx);
    lshn_pnorm<<<CPAT / 8, 256, 0, stream>>>(pat, invp);

    for (int cb = 0; cb < CPAT; cb += CCHUNK) {
        lshn_gemm<<<dim3(CCHUNK / 32, BQ / 128), 256, 0, stream>>>(pat, qn, invp, wptr, sims, cb);
        lshn_topk<<<BQ, 256, 0, stream>>>(sims, rscore, ridx, cb);
    }

    lshn_aggregate<<<BQ, 256, 0, stream>>>(pat, rscore, ridx, wptr, out);
    (void)in_sizes; (void)n_in; (void)out_size; (void)ws_size;
}
